// Model_22634477650234
// MI455X (gfx1250) — matrix operand order repaired and checked by reading, not compiled, not run on hardware
//
#include <hip/hip_runtime.h>
#include <hip/hip_bf16.h>

typedef __attribute__((ext_vector_type(16))) _Float16 v16h;
typedef __attribute__((ext_vector_type(8)))  _Float16 h8;
typedef __attribute__((ext_vector_type(8)))  float    v8f;
typedef __attribute__((ext_vector_type(4)))  int      i4;

#define DIMS   10000
#define KC     64
#define NCHUNK 157            // 157*64 = 10048
#define DPAD   (NCHUNK*KC)
#define NCLS   128
#define NLVL   500
#define NPAD   512
#define BATCHN 256
#define LDA    72             // padded half-stride: 36 dwords -> conflict-free b128
#define MAXT   41.6f
#define MINT   (-19.9f)

// LDS layout (bytes):
//   enc : DPAD halves           = 20096
//   A   : 128 * LDA halves      = 18432
//   B   : 512 * LDA halves      = 73728
//   sb  : 128 u64               = 1024
#define OFF_A  20096
#define OFF_B  (OFF_A + 18432)
#define OFF_SB (OFF_B + 73728)
#define SMEM_BYTES (OFF_SB + 1024)

#if defined(__has_builtin)
#if __has_builtin(__builtin_amdgcn_global_load_async_to_lds_b128)
#define HAVE_ASYNC_LDS 1
#endif
#endif

#ifdef HAVE_ASYNC_LDS
typedef __attribute__((address_space(1))) i4 gi4_t;   // global int4
typedef __attribute__((address_space(3))) i4 li4_t;   // LDS int4
#endif

__device__ __forceinline__ unsigned fkey(float f) {
    unsigned u = __float_as_uint(f);
    return (u & 0x80000000u) ? ~u : (u | 0x80000000u);  // order-preserving map
}

// ---------------------------------------------------------------------------
// Pre-convert W_reg and codebook to f16 images in workspace, zero-padded to
// DPAD columns (and 512 rows for the codebook). Runs once per launch.
// ---------------------------------------------------------------------------
__global__ __launch_bounds__(256)
void cvt_kernel(const float* __restrict__ Wr, const float* __restrict__ cb,
                _Float16* __restrict__ cb16, _Float16* __restrict__ w16)
{
    const int CBE = NPAD * DPAD;   // 512 * 10048
    const int WE  = NCLS * DPAD;   // 128 * 10048
    for (int i = blockIdx.x * blockDim.x + threadIdx.x; i < CBE + WE;
         i += gridDim.x * blockDim.x) {
        if (i < CBE) {
            const int row = i / DPAD, d = i - row * DPAD;
            const float v = (row < NLVL && d < DIMS) ? cb[(size_t)row * DIMS + d] : 0.f;
            cb16[i] = (_Float16)v;
        } else {
            const int j = i - CBE;
            const int row = j / DPAD, d = j - row * DPAD;
            const float v = (d < DIMS) ? Wr[(size_t)row * DIMS + d] : 0.f;
            w16[j] = (_Float16)v;
        }
    }
}

// ---------------------------------------------------------------------------
// Main fused kernel: encode + 256x (128x512x10000 f16 WMMA GEMM) + argmax.
// ---------------------------------------------------------------------------
__global__ __launch_bounds__(512)
void hdc_wmma_kernel(const float* __restrict__ x,
                     const float* __restrict__ Wp,
                     const float* __restrict__ bp,
                     const _Float16* __restrict__ w16,   // [128][DPAD] f16
                     const _Float16* __restrict__ cb16,  // [512][DPAD] f16
                     float* __restrict__ out)
{
    extern __shared__ char smem[];
    _Float16* encS = (_Float16*)smem;
    _Float16* aS   = (_Float16*)(smem + OFF_A);
    _Float16* bS   = (_Float16*)(smem + OFF_B);
    unsigned long long* sb = (unsigned long long*)(smem + OFF_SB);

    const int b    = blockIdx.x;
    const int tid  = threadIdx.x;
    const int lane = tid & 31;
    const int w    = tid >> 5;   // 0..15
    const int wm   = w & 7;      // M slice (16 classes)
    const int wn   = w >> 3;     // N half  (256 levels)

    // ---------- stage 0: encode enc[b, :] into LDS (f16, exact +-1) ----------
    const float* xb = x + (size_t)b * 6;
    const float x0 = xb[0], x1 = xb[1], x2 = xb[2], x3 = xb[3], x4 = xb[4], x5 = xb[5];
    for (int d = tid; d < DPAD; d += 512) {
        _Float16 e = (_Float16)1.0f;
        if (d < DIMS) {
            const float* wr = Wp + (size_t)d * 6;
            float p = x0*wr[0] + x1*wr[1] + x2*wr[2] + x3*wr[3] + x4*wr[4] + x5*wr[5];
            float hv = cosf(p + bp[d]) * sinf(p);
            e = (hv > 0.0f) ? (_Float16)1.0f : (_Float16)(-1.0f);
        }
        encS[d] = e;
    }
    if (tid < NCLS) sb[tid] = 0ull;

    v8f acc[16];
    #pragma unroll
    for (int i = 0; i < 16; ++i)
        acc[i] = v8f{0.f,0.f,0.f,0.f,0.f,0.f,0.f,0.f};

    // ---------- main K loop ----------
    for (int ch = 0; ch < NCHUNK; ++ch) {
        const int dbase = ch * KC;
        __syncthreads();  // previous chunk consumed / enc+sb ready

        // ---- stage B tile: raw f16 async copy, row `tid`, 128B with LDS row pad ----
#ifdef HAVE_ASYNC_LDS
        {
            const _Float16* src = cb16 + (size_t)tid * DPAD + dbase;
            _Float16* dst = bS + tid * LDA;
            #pragma unroll
            for (int i = 0; i < 8; ++i) {
                __builtin_amdgcn_global_load_async_to_lds_b128(
                    (gi4_t*)(src + i * 8), (li4_t*)(dst + i * 8), 0, 0);
            }
        }
#else
        {
            const _Float16* src = cb16 + (size_t)tid * DPAD + dbase;
            _Float16* dst = bS + tid * LDA;
            #pragma unroll
            for (int i = 0; i < 8; ++i)
                *(h8*)(dst + i * 8) = *(const h8*)(src + i * 8);
        }
#endif
        // ---- stage A tile: w16 * enc (exact +-1 sign apply), 2x16B per thread ----
        {
            #pragma unroll
            for (int i = 0; i < 2; ++i) {
                const int f   = tid * 2 + i;
                const int row = f >> 3;        // 0..127
                const int seg = (f & 7) * 8;   // half offset within 64
                h8 v = *(const h8*)(w16 + (size_t)row * DPAD + dbase + seg);
                h8 e = *(const h8*)(encS + dbase + seg);
                *(h8*)(aS + row * LDA + seg) = v * e;
            }
        }
#ifdef HAVE_ASYNC_LDS
#if __has_builtin(__builtin_amdgcn_s_wait_asynccnt)
        __builtin_amdgcn_s_wait_asynccnt(0);
#else
        asm volatile("s_wait_asynccnt 0x0" ::: "memory");
#endif
#endif
        __syncthreads();

        // ---- compute: 2 K-steps of 32, 16 N tiles, B-fragment double buffered ----
        #pragma unroll
        for (int ks = 0; ks < KC; ks += 32) {
            v16h afrag;
            {
                // A 16x32 f16: halves0..7 -> K=ks+(lane>=16?8:0)+0..7; halves8..15 -> +16
                const int mrow = wm * 16 + (lane & 15);
                const int koff = ks + ((lane >= 16) ? 8 : 0);
                h8 lo = *(const h8*)(aS + mrow * LDA + koff);
                h8 hi = *(const h8*)(aS + mrow * LDA + koff + 16);
                #pragma unroll
                for (int i = 0; i < 8; ++i) { afrag[i] = lo[i]; afrag[i + 8] = hi[i]; }
            }
            // B 32x16 f16: halves0..7 -> K = ks + (lane>=16?8:0) + 0..7; halves8..15 -> +16
            const _Float16* bbase =
                bS + (wn * 256 + (lane & 15)) * LDA + ks + ((lane >= 16) ? 8 : 0);
            h8 plo[2], phi[2];
            plo[0] = *(const h8*)(bbase);
            phi[0] = *(const h8*)(bbase + 16);
            #pragma unroll
            for (int nt = 0; nt < 16; ++nt) {
                const int cur = nt & 1;
                if (nt + 1 < 16) {  // prefetch next B fragment while WMMA runs
                    plo[cur ^ 1] = *(const h8*)(bbase + (nt + 1) * 16 * LDA);
                    phi[cur ^ 1] = *(const h8*)(bbase + (nt + 1) * 16 * LDA + 16);
                }
                v16h bfrag;
                #pragma unroll
                for (int i = 0; i < 8; ++i) { bfrag[i] = plo[cur][i]; bfrag[i + 8] = phi[cur][i]; }
                acc[nt] = __builtin_amdgcn_wmma_f32_16x16x32_f16(
                    false, afrag, false, bfrag, (short)0, acc[nt], false, false);
            }
        }
    }

    // ---------- epilogue: argmax over levels per (b, class) ----------
    #pragma unroll
    for (int r = 0; r < 8; ++r) {
        const int m = wm * 16 + r + ((lane >= 16) ? 8 : 0);
        float bestv = -3.4e38f;
        int   bestn = -1;
        #pragma unroll
        for (int nt = 0; nt < 16; ++nt) {
            const int n = wn * 256 + nt * 16 + (lane & 15);
            const float v = acc[nt][r];
            if (n < NLVL && v > bestv) { bestv = v; bestn = n; }
        }
        if (bestn >= 0) {
            unsigned long long p =
                ((unsigned long long)fkey(bestv) << 32) | (unsigned)(~(unsigned)bestn);
            atomicMax(&sb[m], p);   // max value; ties -> smallest index
        }
    }
    __syncthreads();
    if (tid < NCLS) {
        const unsigned long long p = sb[tid];
        const int n = (int)(~(unsigned)(p & 0xFFFFFFFFull));
        const float pred = (float)n * ((MAXT - MINT) / (float)(NLVL - 1)) + MINT;
        out[(size_t)b * NCLS + tid] = pred;
    }
}

extern "C" void kernel_launch(void* const* d_in, const int* in_sizes, int n_in,
                              void* d_out, int out_size, void* d_ws, size_t ws_size,
                              hipStream_t stream) {
    const float* x  = (const float*)d_in[0];   // [256, 6]
    const float* Wp = (const float*)d_in[1];   // [10000, 6]
    const float* bp = (const float*)d_in[2];   // [10000]
    const float* Wr = (const float*)d_in[3];   // [128, 10000]
    const float* cb = (const float*)d_in[4];   // [500, 10000]
    float* out = (float*)d_out;                // [256, 128]

    // workspace: f16 codebook [512][DPAD] then f16 W_reg [128][DPAD] (~12.9 MB)
    _Float16* cb16 = (_Float16*)d_ws;
    _Float16* w16  = cb16 + (size_t)NPAD * DPAD;

    hipLaunchKernelGGL(cvt_kernel, dim3(1024), dim3(256), 0, stream,
                       Wr, cb, cb16, w16);
    hipLaunchKernelGGL(hdc_wmma_kernel, dim3(BATCHN), dim3(512), SMEM_BYTES, stream,
                       x, Wp, bp, w16, cb16, out);
}
